// VanillaRNN_31791347925664
// MI455X (gfx1250) — matrix operand order repaired and checked by reading, not compiled, not run on hardware
//
#include <hip/hip_runtime.h>
#include <hip/hip_bf16.h>

#define DEV __device__ __forceinline__

typedef __bf16 v16bf __attribute__((ext_vector_type(16)));
typedef float  v8f   __attribute__((ext_vector_type(8)));

constexpr int IN = 512;
constexpr int H  = 1024;
constexpr int C  = 1000;
constexpr int NB = 64;    // batch
constexpr int T  = 512;
constexpr int BT = NB * T;

// ---------- helpers ----------

DEV __bf16 f2bf(float f) {
    unsigned u = __builtin_bit_cast(unsigned, f);
    u += 0x7FFFu + ((u >> 16) & 1u);          // round-to-nearest-even
    unsigned short h = (unsigned short)(u >> 16);
    return __builtin_bit_cast(__bf16, h);
}

// A fragment (16x32 bf16, M x K): lane<16 -> row=lane,    K in {k0..k0+15}
//                                 lane>=16 -> row=lane-16, K in {k0+16..k0+31}
DEV v16bf load_frag_a(const __bf16* A, int lda, int row0, int k0, int lane) {
    const __bf16* p = A + (size_t)(row0 + (lane & 15)) * lda + k0 + 16 * (lane >> 4);
    union { uint4 q[2]; v16bf v; } u;
    u.q[0] = *(const uint4*)(p);
    u.q[1] = *(const uint4*)(p + 8);
    return u.v;
}

// B fragment (32x16 bf16, K x N) from [n][k]-row storage: lane holds col n = n0+(lane&15),
// 16 contiguous K values starting at k0 + 16*(lane>>4)
DEV v16bf load_frag_b(const __bf16* Bm, int ldb, int n0, int k0, int lane) {
    const __bf16* p = Bm + (size_t)(n0 + (lane & 15)) * ldb + k0 + 16 * (lane >> 4);
    union { uint4 q[2]; v16bf v; } u;
    u.q[0] = *(const uint4*)(p);
    u.q[1] = *(const uint4*)(p + 8);
    return u.v;
}

// B fragment from pre-swizzled fragment-major storage: frag(nt,kt) is a contiguous
// 1KB block, lane-ordered (32B per lane) -> perfectly coalesced L2 reads.
DEV v16bf load_frag_b_sw(const __bf16* fragBase) {
    union { uint4 q[2]; v16bf v; } u;
    u.q[0] = *(const uint4*)(fragBase);
    u.q[1] = *(const uint4*)(fragBase + 8);
    return u.v;
}

DEV v8f wmma_bf16(v16bf a, v16bf b, v8f c) {
    return __builtin_amdgcn_wmma_f32_16x16x32_bf16(false, a, false, b, (short)0, c,
                                                   false, false);
}

// CDNA5 async copy: per-lane 16B global -> LDS, tracked by ASYNCcnt.
DEV void async_copy_b128(void* lds_dst, const void* gsrc) {
    unsigned ldsa = (unsigned)(uintptr_t)lds_dst;          // low 32 bits = LDS offset
    unsigned long long ga = (unsigned long long)(uintptr_t)gsrc;
    asm volatile("global_load_async_to_lds_b128 %0, %1, off"
                 :: "v"(ldsa), "v"(ga) : "memory");
}
DEV void wait_async0() {
    asm volatile("s_wait_asynccnt 0x0" ::: "memory");
}

// ---------- phase 0: Wh f32 -> bf16, pre-swizzled into fragment order ----------
// one thread per (nt, kt, lane); 64*32*32 = 65536 threads.
__global__ __launch_bounds__(256) void wh_convert_kernel(
        const float* __restrict__ Wh, __bf16* __restrict__ Whb) {
    const int idx  = blockIdx.x * 256 + threadIdx.x;
    const int lane = idx & 31;
    const int kt   = (idx >> 5) & 31;
    const int nt   = idx >> 10;                 // 0..63
    const int n = nt * 16 + (lane & 15);
    const int k = kt * 32 + 16 * (lane >> 4);
    const float* src = Wh + (size_t)n * H + k;
    __bf16* dst = Whb + (size_t)idx * 16;
#pragma unroll
    for (int j = 0; j < 16; ++j) dst[j] = f2bf(src[j]);
}

// ---------- phase 1: xh[t][b][n] = bf16( x @ Wx^T + bx ) ----------
// grid (512, 8), block 256.  Block tile: 64 (M=bt) x 128 (N=h), K-step 32.
__global__ __launch_bounds__(256) void xh_gemm_kernel(
        const float* __restrict__ x, const float* __restrict__ Wx,
        const float* __restrict__ bx, __bf16* __restrict__ xh2) {
    __shared__ __bf16 As[64 * 32];    // [m][k]
    __shared__ __bf16 Bs[128 * 32];   // [n][k]
    const int tid = threadIdx.x, lane = tid & 31, w = tid >> 5;
    const int wm = w & 3, wn = w >> 2;                // 4x2 wave grid
    const int row0 = blockIdx.x * 64, col0 = blockIdx.y * 128;
    v8f acc[4] = {};
    for (int k0 = 0; k0 < IN; k0 += 32) {
        {   // stage x tile, f32 -> bf16 (8 contiguous per thread)
            const int m = tid >> 2, kk = (tid & 3) * 8;
            const float* xp = x + (size_t)(row0 + m) * IN + k0 + kk;
            __bf16* ap = As + m * 32 + kk;
#pragma unroll
            for (int i = 0; i < 8; ++i) ap[i] = f2bf(xp[i]);
        }
        {   // stage Wx tile (16 contiguous per thread)
            const int n = tid >> 1, kk = (tid & 1) * 16;
            const float* wp = Wx + (size_t)(col0 + n) * IN + k0 + kk;
            __bf16* bp = Bs + n * 32 + kk;
#pragma unroll
            for (int i = 0; i < 16; ++i) bp[i] = f2bf(wp[i]);
        }
        __syncthreads();
        v16bf a = load_frag_a(As, 32, wm * 16, 0, lane);
#pragma unroll
        for (int j = 0; j < 4; ++j) {
            v16bf b = load_frag_b(Bs, 32, wn * 64 + j * 16, 0, lane);
            acc[j] = wmma_bf16(a, b, acc[j]);
        }
        __syncthreads();
    }
#pragma unroll
    for (int j = 0; j < 4; ++j) {
        const int n = col0 + wn * 64 + j * 16 + (lane & 15);
        const float bias = bx[n];
#pragma unroll
        for (int r = 0; r < 8; ++r) {
            const int m = row0 + wm * 16 + r + 8 * (lane >> 4);   // bt index
            const int b = m >> 9, t = m & (T - 1);
            xh2[((size_t)t * NB + b) * H + n] = f2bf(acc[j][r] + bias);
        }
    }
}

// ---------- phase 2: batch-partitioned scan, NO inter-block sync ----------
// grid 4, block 512 (16 waves). Block g owns batch rows [16g, 16g+16).
// h state double-buffered in LDS; Wh streamed from L2 (pre-swizzled bf16);
// next step's xh slab prefetched to LDS with async-to-LDS copies.
__global__ __launch_bounds__(512) void rnn_scan_kernel(
        const __bf16* __restrict__ Whb, const __bf16* __restrict__ xh2,
        __bf16* __restrict__ hout) {
    __shared__ __bf16 hbuf[2][16 * 1024];   // 2 x 32 KB
    __shared__ __bf16 xhs[2][16 * 1024];    // 2 x 32 KB
    const int tid = threadIdx.x, lane = tid & 31, w = tid >> 5;  // 16 waves
    const int g = blockIdx.x;
    const int nt0 = w * 4;                   // wave's 4 N-tiles (64 cols)
    // this wave's 4 fragment columns in the swizzled Wh (frag stride = 512 elems)
    const __bf16* wcol = Whb + ((size_t)nt0 * 32 + lane) * 16;
    // zero h state
    for (int i = tid; i < 16 * 1024; i += 512) hbuf[0][i] = f2bf(0.f);
    // prefetch xh slab for t = 0
    {
        const __bf16* src = xh2 + (size_t)16 * g * H;     // t = 0
#pragma unroll
        for (int i = 0; i < 4; ++i) {
            const int c = i * 512 + tid;                  // 16-byte chunk index
            async_copy_b128((char*)xhs[0] + c * 16, (const char*)src + c * 16);
        }
        wait_async0();
    }
    __syncthreads();
    int cur = 0;
    for (int t = 0; t < T; ++t) {
        // issue async prefetch of next xh slab; overlaps the whole K-loop
        {
            const int tn = (t + 1 < T) ? (t + 1) : t;
            const __bf16* src = xh2 + ((size_t)tn * NB + 16 * g) * H;
            char* dst = (char*)xhs[(t + 1) & 1];
#pragma unroll
            for (int i = 0; i < 4; ++i) {
                const int c = i * 512 + tid;
                async_copy_b128(dst + c * 16, (const char*)src + c * 16);
            }
        }
        const __bf16* hcur = hbuf[cur];
        v8f acc[4] = {};
#pragma unroll 4
        for (int kt = 0; kt < 32; ++kt) {
            v16bf a = load_frag_a(hcur, H, 0, kt * 32, lane);
            const __bf16* fb = wcol + (size_t)kt * 512;
#pragma unroll
            for (int j = 0; j < 4; ++j) {
                v16bf b = load_frag_b_sw(fb + (size_t)j * 32 * 512);
                acc[j] = wmma_bf16(a, b, acc[j]);
            }
        }
        // epilogue: h_new = tanh(acc + xh_t), into other LDS buffer
        const __bf16* xs = xhs[t & 1];
        __bf16* hn = hbuf[cur ^ 1];
#pragma unroll
        for (int j = 0; j < 4; ++j) {
            const int n = (nt0 + j) * 16 + (lane & 15);
#pragma unroll
            for (int r = 0; r < 8; ++r) {
                const int m = r + 8 * (lane >> 4);
                const float v = acc[j][r] + (float)xs[m * H + n];
                hn[m * H + n] = f2bf(tanhf(v));
            }
        }
        wait_async0();        // own wave's async copies done
        __syncthreads();      // everyone's copies + h writes visible
        cur ^= 1;
    }
    // write final h rows to global
    for (int i = tid; i < 16 * 1024; i += 512) {
        const int m = i >> 10, n = i & (H - 1);
        hout[(size_t)(16 * g + m) * H + n] = hbuf[cur][i];
    }
}

// ---------- phase 3: logits = hT @ Wy^T + by ----------
// grid 8, block 256. Block tile 64 (M) x 128 (N, padded to 1024), K=1024.
__global__ __launch_bounds__(256) void logits_kernel(
        const __bf16* __restrict__ h, const float* __restrict__ Wy,
        const float* __restrict__ by, float* __restrict__ out) {
    __shared__ __bf16 Bs[128 * 32];
    const int tid = threadIdx.x, lane = tid & 31, w = tid >> 5;
    const int wm = w & 3, wn = w >> 2;
    const int col0 = blockIdx.x * 128;
    v8f acc[4] = {};
    for (int k0 = 0; k0 < H; k0 += 32) {
        {
            const int n = tid >> 1, kk = (tid & 1) * 16;
            const int gn = col0 + n;
            __bf16* bp = Bs + n * 32 + kk;
            if (gn < C) {
                const float* wp = Wy + (size_t)gn * H + k0 + kk;
#pragma unroll
                for (int i = 0; i < 16; ++i) bp[i] = f2bf(wp[i]);
            } else {
#pragma unroll
                for (int i = 0; i < 16; ++i) bp[i] = f2bf(0.f);
            }
        }
        __syncthreads();
        v16bf a = load_frag_a(h, H, wm * 16, k0, lane);
#pragma unroll
        for (int j = 0; j < 4; ++j) {
            v16bf b = load_frag_b(Bs, 32, wn * 64 + j * 16, 0, lane);
            acc[j] = wmma_bf16(a, b, acc[j]);
        }
        __syncthreads();
    }
#pragma unroll
    for (int j = 0; j < 4; ++j) {
        const int n = col0 + wn * 64 + j * 16 + (lane & 15);
        if (n < C) {
            const float bias = by[n];
#pragma unroll
            for (int r = 0; r < 8; ++r) {
                const int m = wm * 16 + r + 8 * (lane >> 4);
                out[(size_t)m * C + n] = acc[j][r] + bias;
            }
        }
    }
}

// ---------- host ----------
extern "C" void kernel_launch(void* const* d_in, const int* in_sizes, int n_in,
                              void* d_out, int out_size, void* d_ws, size_t ws_size,
                              hipStream_t stream) {
    const float* x  = (const float*)d_in[0];
    const float* Wx = (const float*)d_in[1];
    const float* bx = (const float*)d_in[2];
    const float* Wh = (const float*)d_in[3];
    const float* Wy = (const float*)d_in[4];
    const float* by = (const float*)d_in[5];
    float* out = (float*)d_out;

    char* ws = (char*)d_ws;
    __bf16* xh2 = (__bf16*)ws;                                   // [T][NB][H] bf16 = 64 MB
    size_t off = (size_t)BT * H * sizeof(__bf16);
    __bf16* Whb = (__bf16*)(ws + off); off += (size_t)H * H * sizeof(__bf16);   // 2 MB
    __bf16* hfin = (__bf16*)(ws + off);                          // [NB][H] bf16 = 128 KB

    wh_convert_kernel<<<(H * H / 16) / 256, 256, 0, stream>>>(Wh, Whb);
    xh_gemm_kernel<<<dim3(BT / 64, H / 128), 256, 0, stream>>>(x, Wx, bx, xh2);
    rnn_scan_kernel<<<NB / 16, 512, 0, stream>>>(Whb, xh2, hfin);
    logits_kernel<<<H / 128, 256, 0, stream>>>(hfin, Wy, by, out);
}
